// WaveletAttention_17394617548909
// MI455X (gfx1250) — matrix operand order repaired and checked by reading, not compiled, not run on hardware
//
#include <hip/hip_runtime.h>

#define C_ISQ2 0.70710678118654752440f

typedef __attribute__((ext_vector_type(16))) _Float16 v16h;
typedef __attribute__((ext_vector_type(8)))  float    v8f;

// ---------------------------------------------------------------------------
// Zero a small scratch region (logit accumulators) — required every launch.
// ---------------------------------------------------------------------------
__global__ void wa_zero_kernel(float* __restrict__ p, int n) {
    int i = blockIdx.x * blockDim.x + threadIdx.x;
    if (i < n) p[i] = 0.0f;
}

// ---------------------------------------------------------------------------
// One 2D Haar analysis level. Each thread produces one (a,h,v,d) quad from a
// 2x2 input block. Rounding order matches the reference (two INV_SQRT2 muls).
// src: [B, 2*Hh, 2*Wh] row-major; outputs: [B, Hh*Wh] each.
// ---------------------------------------------------------------------------
__global__ void wa_dwt_kernel(const float* __restrict__ src,
                              float* __restrict__ a, float* __restrict__ h,
                              float* __restrict__ v, float* __restrict__ d,
                              int Hh, int Wh, int total) {
    int idx = blockIdx.x * blockDim.x + threadIdx.x;
    if (idx >= total) return;
    int n = Hh * Wh;
    int b = idx / n;
    int r = idx - b * n;
    int i = r / Wh;
    int j = r - i * Wh;
    int W = Wh * 2;
    const float* base = src + (size_t)b * (size_t)(Hh * 2) * (size_t)W
                            + (size_t)(2 * i) * (size_t)W + (size_t)(2 * j);
    float x00 = base[0], x01 = base[1];
    float x10 = base[W], x11 = base[W + 1];
    float lo0 = (x00 + x01) * C_ISQ2, hi0 = (x00 - x01) * C_ISQ2;
    float lo1 = (x10 + x11) * C_ISQ2, hi1 = (x10 - x11) * C_ISQ2;
    a[idx] = (lo0 + lo1) * C_ISQ2;
    h[idx] = (lo0 - lo1) * C_ISQ2;
    v[idx] = (hi0 + hi1) * C_ISQ2;
    d[idx] = (hi0 - hi1) * C_ISQ2;
}

// ---------------------------------------------------------------------------
// Gate GEMV via WMMA:  logits[m][n] += sum_k A[m][k] * W[k][n]
//   A: [16, K] row-major f32 (wavelet coefficients, M = batch)
//   W: [K, NCOLS] row-major f32 (NCOLS = 1 or 3, padded to N=16 with zeros)
// One wave per block; each block owns a 4096-wide K chunk and runs
// V_WMMA_F32_16X16X32_F16 over it, then atomically reduces into logits.
//
// A-operand packing per ISA 7.12.2 "16-bit A-Matrix 16x32":
//   lanes 0-15 -> M = lane; half-pair g: K = (g>=4?16:0) + 2*(g&3)
//   lanes 16-31 -> same M, K offset +8
// B-operand packing per the dense B layout (ISA 7.12.4 pattern at K=32):
//   lane%16 = column n; same K order as A: element s -> K = 8*hi + s (s<8),
//   16 + 8*hi + (s-8) (s>=8).
// C/D layout: lanes 0-15 VGPR r -> (M=r, N=lane); lanes 16-31 -> (M=8+r).
// ---------------------------------------------------------------------------
template <int NCOLS>
__global__ __launch_bounds__(32)
void wa_gate_wmma_kernel(const float* __restrict__ A,
                         const float* __restrict__ W,
                         int K,
                         float* __restrict__ logits) {
    const int  lane   = threadIdx.x & 31;
    const int  m      = lane & 15;
    const int  hi     = lane >> 4;          // 0 or 1
    const bool active = (m < NCOLS);
    const int  CHUNK  = 4096;
    const int  k0     = blockIdx.x * CHUNK;

    const float* ap = A + (size_t)m * (size_t)K + (size_t)k0;
    // Column pointer for this lane's B column; inactive lanes get a safe base
    // (they never load through it).
    const float* wp = W + (size_t)(k0 + hi * 8) * NCOLS + (active ? m : 0);

    v8f acc = {};
    for (int t = 0; t < CHUNK / 32; ++t) {
        __builtin_prefetch(ap + 96, 0, 1);   // A stream ~3 tiles ahead

        // ---- pack A tile (16x32 f16) ----
        v16h av;
#pragma unroll
        for (int g = 0; g < 8; ++g) {
            const int kb = ((g >= 4) ? 16 : 0) + (hi ? 8 : 0) + 2 * (g & 3);
            av[2 * g]     = (_Float16)ap[kb];
            av[2 * g + 1] = (_Float16)ap[kb + 1];
        }

        // ---- pack B tile (32x16 f16): one predicated region per tile ----
        v16h bv = {};
        if (active) {
#pragma unroll
            for (int j = 0; j < 16; ++j)
                bv[j] = (_Float16)wp[(j + (j >= 8 ? 8 : 0)) * NCOLS];
        }
        // EXEC is fully restored here; WMMA requires all lanes active.

        acc = __builtin_amdgcn_wmma_f32_16x16x32_f16(
            /*neg_a=*/false, av, /*neg_b=*/false, bv,
            /*c_mod=*/(short)0, acc, /*reuse_a=*/false, /*reuse_b=*/false);

        ap += 32;
        wp += 32 * NCOLS;
    }

    // ---- reduce D column m (valid columns only) into logits ----
    if (active) {
#pragma unroll
        for (int r = 0; r < 8; ++r) {
            const int row = r + hi * 8;      // batch index
            unsafeAtomicAdd(&logits[row * NCOLS + m], acc[r]);
        }
    }
}

// ---------------------------------------------------------------------------
// Turn accumulated logits into gate values.
// logits layout: [0..15] approx logit; +16 + L*48: level-L [16][3] logits.
// gates  layout: [0..15] aw (sigmoid);  +16 + L*48: level-L [16][3] softmax.
// ---------------------------------------------------------------------------
__global__ void wa_finalize_gates_kernel(const float* __restrict__ logits,
                                         const float* __restrict__ ba,
                                         const float* __restrict__ bd1,
                                         const float* __restrict__ bd2,
                                         const float* __restrict__ bd3,
                                         float* __restrict__ gates) {
    int b = threadIdx.x;
    if (b >= 16) return;
    float la = logits[b] + ba[0];
    gates[b] = 1.0f / (1.0f + expf(-la));

    const float* bds[3] = {bd1, bd2, bd3};
#pragma unroll
    for (int L = 0; L < 3; ++L) {
        const float* lg = logits + 16 + L * 48 + b * 3;
        float l0 = lg[0] + bds[L][0];
        float l1 = lg[1] + bds[L][1];
        float l2 = lg[2] + bds[L][2];
        float mx = fmaxf(l0, fmaxf(l1, l2));
        float e0 = expf(l0 - mx);
        float e1 = expf(l1 - mx);
        float e2 = expf(l2 - mx);
        float inv = 1.0f / (e0 + e1 + e2);
        float* g = gates + 16 + L * 48 + b * 3;
        g[0] = e0 * inv;
        g[1] = e1 * inv;
        g[2] = e2 * inv;
    }
}

// ---------------------------------------------------------------------------
// One 2D Haar synthesis level with per-batch gates.
// a,h,v,d: [B, Hh*Wh]; out: [B, 2*Hh, 2*Wh].
// a is multiplied by ga[b] iff use_ga != 0 (only the coarsest level).
// h,v,d are multiplied by gd[b*3 + {0,1,2}].
// ---------------------------------------------------------------------------
__global__ void wa_idwt_kernel(const float* __restrict__ a,
                               const float* __restrict__ h,
                               const float* __restrict__ v,
                               const float* __restrict__ d,
                               const float* __restrict__ ga, int use_ga,
                               const float* __restrict__ gd,
                               float* __restrict__ out,
                               int Hh, int Wh, int total) {
    int idx = blockIdx.x * blockDim.x + threadIdx.x;
    if (idx >= total) return;
    int n = Hh * Wh;
    int b = idx / n;
    int r = idx - b * n;
    int i = r / Wh;
    int j = r - i * Wh;

    float wa = use_ga ? ga[b] : 1.0f;
    float wh = gd[b * 3 + 0];
    float wv = gd[b * 3 + 1];
    float wd = gd[b * 3 + 2];

    float A = a[idx] * wa;
    float H = h[idx] * wh;
    float V = v[idx] * wv;
    float D = d[idx] * wd;

    float l0 = (A + H) * C_ISQ2, l1 = (A - H) * C_ISQ2;
    float h0 = (V + D) * C_ISQ2, h1 = (V - D) * C_ISQ2;

    int W = Wh * 2;
    float* o = out + (size_t)b * (size_t)(Hh * 2) * (size_t)W
                   + (size_t)(2 * i) * (size_t)W + (size_t)(2 * j);
    o[0]     = (l0 + h0) * C_ISQ2;
    o[1]     = (l0 - h0) * C_ISQ2;
    o[W]     = (l1 + h1) * C_ISQ2;
    o[W + 1] = (l1 - h1) * C_ISQ2;
}

// ---------------------------------------------------------------------------
// Host-side orchestration.
// ---------------------------------------------------------------------------
extern "C" void kernel_launch(void* const* d_in, const int* in_sizes, int n_in,
                              void* d_out, int out_size, void* d_ws, size_t ws_size,
                              hipStream_t stream) {
    (void)in_sizes; (void)n_in; (void)out_size; (void)ws_size;

    const float* x   = (const float*)d_in[0];
    const float* Wa  = (const float*)d_in[1];
    const float* ba  = (const float*)d_in[2];
    const float* Wd1 = (const float*)d_in[3];  // coarsest (level-3) details
    const float* bd1 = (const float*)d_in[4];
    const float* Wd2 = (const float*)d_in[5];
    const float* bd2 = (const float*)d_in[6];
    const float* Wd3 = (const float*)d_in[7];  // finest (level-1) details
    const float* bd3 = (const float*)d_in[8];
    float* out = (float*)d_out;
    float* ws  = (float*)d_ws;

    const int B  = 16;
    const int n1 = 512 * 512;   // level-1 coeff count per image
    const int n2 = 256 * 256;
    const int n3 = 128 * 128;

    size_t off = 0;
    float* A1 = ws + off; off += (size_t)B * n1;
    float* H1 = ws + off; off += (size_t)B * n1;
    float* V1 = ws + off; off += (size_t)B * n1;
    float* D1 = ws + off; off += (size_t)B * n1;
    float* A2 = ws + off; off += (size_t)B * n2;
    float* H2 = ws + off; off += (size_t)B * n2;
    float* V2 = ws + off; off += (size_t)B * n2;
    float* D2 = ws + off; off += (size_t)B * n2;
    float* A3 = ws + off; off += (size_t)B * n3;
    float* H3 = ws + off; off += (size_t)B * n3;
    float* V3 = ws + off; off += (size_t)B * n3;
    float* D3 = ws + off; off += (size_t)B * n3;
    float* logits = ws + off; off += 160;   // 16 approx + 3*48 detail logits
    float* gates  = ws + off; off += 160;   // 16 aw     + 3*48 softmax gates

    const int T = 256;
    int t1 = B * n1, t2 = B * n2, t3 = B * n3;

    // 0) zero logit accumulators (deterministic every call)
    wa_zero_kernel<<<1, 256, 0, stream>>>(logits, 160);

    // 1) forward DWT, 3 levels
    wa_dwt_kernel<<<(t1 + T - 1) / T, T, 0, stream>>>(x,  A1, H1, V1, D1, 512, 512, t1);
    wa_dwt_kernel<<<(t2 + T - 1) / T, T, 0, stream>>>(A1, A2, H2, V2, D2, 256, 256, t2);
    wa_dwt_kernel<<<(t3 + T - 1) / T, T, 0, stream>>>(A2, A3, H3, V3, D3, 128, 128, t3);

    // 2) gate logits via WMMA GEMVs (K multiples of 4096 throughout)
    float* lgA = logits;
    float* lg1 = logits + 16;        // level-3 (coarsest) detail logits [16][3]
    float* lg2 = logits + 16 + 48;
    float* lg3 = logits + 16 + 96;   // level-1 (finest)

    wa_gate_wmma_kernel<1><<<n3 / 4096, 32, 0, stream>>>(A3, Wa, n3, lgA);

    wa_gate_wmma_kernel<3><<<n3 / 4096, 32, 0, stream>>>(H3, Wd1 + (size_t)0 * n3 * 3, n3, lg1);
    wa_gate_wmma_kernel<3><<<n3 / 4096, 32, 0, stream>>>(V3, Wd1 + (size_t)1 * n3 * 3, n3, lg1);
    wa_gate_wmma_kernel<3><<<n3 / 4096, 32, 0, stream>>>(D3, Wd1 + (size_t)2 * n3 * 3, n3, lg1);

    wa_gate_wmma_kernel<3><<<n2 / 4096, 32, 0, stream>>>(H2, Wd2 + (size_t)0 * n2 * 3, n2, lg2);
    wa_gate_wmma_kernel<3><<<n2 / 4096, 32, 0, stream>>>(V2, Wd2 + (size_t)1 * n2 * 3, n2, lg2);
    wa_gate_wmma_kernel<3><<<n2 / 4096, 32, 0, stream>>>(D2, Wd2 + (size_t)2 * n2 * 3, n2, lg2);

    wa_gate_wmma_kernel<3><<<n1 / 4096, 32, 0, stream>>>(H1, Wd3 + (size_t)0 * n1 * 3, n1, lg3);
    wa_gate_wmma_kernel<3><<<n1 / 4096, 32, 0, stream>>>(V1, Wd3 + (size_t)1 * n1 * 3, n1, lg3);
    wa_gate_wmma_kernel<3><<<n1 / 4096, 32, 0, stream>>>(D1, Wd3 + (size_t)2 * n1 * 3, n1, lg3);

    // 3) sigmoid / softmax
    wa_finalize_gates_kernel<<<1, 16, 0, stream>>>(logits, ba, bd1, bd2, bd3, gates);

    float* aw = gates;
    float* g1 = gates + 16;
    float* g2 = gates + 64;
    float* g3 = gates + 112;

    // 4) gated inverse DWT, reusing freed approx buffers as reconstruction
    float* rec256 = A2;   // A2 no longer needed
    float* rec512 = A1;   // A1 no longer needed
    wa_idwt_kernel<<<(t3 + T - 1) / T, T, 0, stream>>>(A3,     H3, V3, D3, aw, 1, g1, rec256, 128, 128, t3);
    wa_idwt_kernel<<<(t2 + T - 1) / T, T, 0, stream>>>(rec256, H2, V2, D2, aw, 0, g2, rec512, 256, 256, t2);
    wa_idwt_kernel<<<(t1 + T - 1) / T, T, 0, stream>>>(rec512, H1, V1, D1, aw, 0, g3, out,    512, 512, t1);
}
